// CSA_61452392071719
// MI455X (gfx1250) — compile-verified, matrix operand order checked by reading, not run on hardware
//
#include <hip/hip_runtime.h>
#include <hip/hip_bf16.h>

typedef __attribute__((ext_vector_type(16))) __bf16 v16bf;
typedef __attribute__((ext_vector_type(8)))  float  v8f;
typedef __attribute__((ext_vector_type(8)))  unsigned short us8;

// ---------------- helpers ----------------

__device__ __forceinline__ unsigned short f2bf(float f) {
    unsigned int u = __float_as_uint(f);
    unsigned int r = u + 0x7FFFu + ((u >> 16) & 1u);   // round-to-nearest-even
    return (unsigned short)(r >> 16);
}

// ---------------- 1. convert x (f32, K=120) -> bf16 padded to K=128 ----------------

__global__ void cvt_pad128(const float* __restrict__ X, unsigned short* __restrict__ Y,
                           size_t total /* rows*128 */, int K /*120*/) {
    size_t idx = (size_t)blockIdx.x * blockDim.x + threadIdx.x;
    if (idx >= total) return;
    int    c = (int)(idx & 127);
    size_t r = idx >> 7;
    float  v = (c < K) ? X[r * (size_t)K + c] : 0.f;
    Y[idx] = f2bf(v);
}

// ---------------- 2. swizzle weights into WMMA bf16 B-fragment order ----------------
// out layout: [ntile][kt(4)][lane(32)][elem(16)]  (bf16)
// B fragment mapping (wave32, 16-bit B 32x16):
//   lane l: n = (l&15), hi = l>>4
//   elem e: i=e>>1, k = kt*32 + (i<4?0:16) + 8*hi + 2*(i&3) + (e&1)

__global__ void swizzle_w(const float* __restrict__ W, int K, int N, int total,
                          unsigned short* __restrict__ out) {
    int idx = blockIdx.x * blockDim.x + threadIdx.x;
    if (idx >= total) return;
    int e     = idx & 15;
    int lane  = (idx >> 4) & 31;
    int kt    = (idx >> 9) & 3;
    int ntile = idx >> 11;
    int n  = (ntile << 4) + (lane & 15);
    int hi = lane >> 4;
    int i  = e >> 1;
    int k  = (kt << 5) + ((i < 4) ? 0 : 16) + (hi << 3) + ((i & 3) << 1) + (e & 1);
    float v = (n < N && k < K) ? W[(size_t)k * N + n] : 0.f;
    out[idx] = f2bf(v);
}

// ---------------- 3. WMMA GEMM: D[M x ldd] = A[M x 128](bf16) * W + bias ----------------
// 8 waves/block, each wave owns one 16-row M-tile; A fragments loaded ONCE into
// registers; pre-swizzled B staged through LDS in 32KB chunks; wave loops over all
// N-tiles -> A traffic is read exactly once from memory (vs once per N-tile).

#define GCHUNK 8   // N-tiles per LDS chunk (8 * 2048 bf16 = 32 KB)

template <int NTILES>
__global__ void wmma_gemm2(const unsigned short* __restrict__ A,
                           const unsigned short* __restrict__ Bs,
                           const float* __restrict__ bias,
                           float* __restrict__ D, int N, int ldd) {
    __shared__ unsigned short sB[GCHUNK * 2048];

    int lane  = threadIdx.x & 31;
    int wave  = threadIdx.x >> 5;
    int mtile = (blockIdx.x << 3) + wave;
    int hi    = lane >> 4;
    int mrow  = (mtile << 4) + (lane & 15);

    const unsigned short* arow = A + ((size_t)mrow << 7) + (hi << 3);

    // prefetch the next block's A rows (speculative; dropped if OOB)
    __builtin_prefetch(arow + (size_t)(128 << 7), 0, 3);

    union frag { us8 u[2]; v16bf v; };
    frag af[4];
#pragma unroll
    for (int kt = 0; kt < 4; ++kt) {
        af[kt].u[0] = *(const us8*)(arow + (kt << 5));
        af[kt].u[1] = *(const us8*)(arow + (kt << 5) + 16);
    }

    constexpr int NCH = (NTILES + GCHUNK - 1) / GCHUNK;
#pragma unroll
    for (int c = 0; c < NCH; ++c) {
        constexpr int CH = GCHUNK;
        int nt0    = c * CH;
        int nchunk = (NTILES - nt0 < CH) ? (NTILES - nt0) : CH;

        __syncthreads();   // protect LDS reuse across chunks
        // cooperative stage of this weight chunk into LDS (128-bit moves)
        int tot8 = nchunk << 8;   // nchunk * 2048 shorts / 8
        for (int i = threadIdx.x; i < tot8; i += 256)
            *(us8*)(sB + (i << 3)) = *(const us8*)(Bs + ((size_t)nt0 << 11) + (i << 3));
        __syncthreads();

#pragma unroll
        for (int nt = 0; nt < CH; ++nt) {
            if (nt0 + nt >= NTILES) break;
            if (nt >= nchunk) break;
            const unsigned short* bp = sB + (nt << 11) + (lane << 4);
            v8f acc = {0.f, 0.f, 0.f, 0.f, 0.f, 0.f, 0.f, 0.f};
#pragma unroll
            for (int kt = 0; kt < 4; ++kt) {
                frag bf_;
                bf_.u[0] = *(const us8*)(bp + (kt << 9));
                bf_.u[1] = *(const us8*)(bp + (kt << 9) + 8);
                acc = __builtin_amdgcn_wmma_f32_16x16x32_bf16(
                    false, af[kt].v, false, bf_.v, (short)0, acc, false, false);
            }
            int ntile = nt0 + nt;
            int n = (ntile << 4) + (lane & 15);
            if (n < N) {
                float bv = bias[n];
#pragma unroll
                for (int j = 0; j < 8; ++j) {
                    int row = (mtile << 4) + (hi << 3) + j;
                    D[(size_t)row * ldd + n] = acc[j] + bv;
                }
            }
        }
    }
}

// ---------------- 4. pooled query (avg first 30 ch, max last 30 ch) ----------------
// qp layout: [branch][b][head(3)][n(8)][d(20)]  f32
// pooling over strided blocks: h = hsp*(H/Hsp)+j, w = wsp*(W/Wsp)+k ; roll folded in.

__global__ void pool_q(const float* __restrict__ qkv, float* __restrict__ qp) {
    int branch = blockIdx.y;
    int bx = blockIdx.x;
    int n  = bx & 7;
    int ch = (bx >> 3) % 60;
    int b  = bx / 480;
    int Hsp = branch ? 4 : 2, Wsp = branch ? 2 : 4;
    int sh  = branch ? 2 : 1, sw  = branch ? 1 : 2;
    int Hd = 256 / Hsp, Wd = 256 / Wsp;
    int hsp = n / Wsp, wsp = n % Wsp;
    int gch = branch * 60 + ch;

    float s = 0.f, mx = -3.4e38f;
    for (int e = threadIdx.x; e < 8192; e += 256) {
        int j = e / Wd, kk = e % Wd;
        int h = hsp * Hd + j, w = wsp * Wd + kk;
        int oh = (h + sh) & 255, ow = (w + sw) & 255;
        float v = qkv[((((size_t)b << 16) + ((size_t)oh << 8) + ow)) * 360 + gch];
        s += v;
        mx = fmaxf(mx, v);
    }
    __shared__ float ssum[256], smax[256];
    int tid = threadIdx.x;
    ssum[tid] = s; smax[tid] = mx;
    __syncthreads();
    for (int st = 128; st > 0; st >>= 1) {
        if (tid < st) {
            ssum[tid] += ssum[tid + st];
            smax[tid] = fmaxf(smax[tid], smax[tid + st]);
        }
        __syncthreads();
    }
    if (tid == 0) {
        float r = (ch < 30) ? ssum[0] * (1.f / 8192.f) : smax[0];
        qp[((((size_t)branch * 4 + b) * 3 + ch / 20) * 8 + n) * 20 + (ch % 20)] = r;
    }
}

// ---------------- 5. DynamicPosBias MLP + rel-pos gather ----------------
// btab layout: [branch][head(3)][n(8)][m(8)]  f32

__device__ __forceinline__ void ln_relu7(float* h, const float* g, const float* be) {
    float m = 0.f;
#pragma unroll
    for (int i = 0; i < 7; ++i) m += h[i];
    m *= (1.f / 7.f);
    float v = 0.f;
#pragma unroll
    for (int i = 0; i < 7; ++i) { float d = h[i] - m; v += d * d; }
    v *= (1.f / 7.f);
    float inv = rsqrtf(v + 1e-5f);
#pragma unroll
    for (int i = 0; i < 7; ++i) h[i] = fmaxf((h[i] - m) * inv * g[i] + be[i], 0.f);
}

__global__ void dyn_bias(const float* __restrict__ w0, const float* __restrict__ b0,
                         const float* __restrict__ g1, const float* __restrict__ be1,
                         const float* __restrict__ w1, const float* __restrict__ b1,
                         const float* __restrict__ g2, const float* __restrict__ be2,
                         const float* __restrict__ w2, const float* __restrict__ b2,
                         const float* __restrict__ g3, const float* __restrict__ be3,
                         const float* __restrict__ w3, const float* __restrict__ b3,
                         float* __restrict__ btab) {
    __shared__ float mlp[2][21][3];
    int t = threadIdx.x;
    if (t < 42) {
        int br = t / 21, off = t % 21;
        int Hsp = br ? 4 : 2, Wsp = br ? 2 : 4;
        int dh = off / (2 * Wsp - 1) - (Hsp - 1);
        int dw = off % (2 * Wsp - 1) - (Wsp - 1);
        float h[7], h2[7];
        const float* W0 = w0 + br * 14;
#pragma unroll
        for (int j = 0; j < 7; ++j)
            h[j] = (float)dh * W0[j] + (float)dw * W0[7 + j] + b0[br * 7 + j];
        ln_relu7(h, g1 + br * 7, be1 + br * 7);
        for (int j = 0; j < 7; ++j) {
            float s = b1[br * 7 + j];
            for (int i = 0; i < 7; ++i) s += h[i] * w1[br * 49 + i * 7 + j];
            h2[j] = s;
        }
        for (int j = 0; j < 7; ++j) h[j] = h2[j];
        ln_relu7(h, g2 + br * 7, be2 + br * 7);
        for (int j = 0; j < 7; ++j) {
            float s = b2[br * 7 + j];
            for (int i = 0; i < 7; ++i) s += h[i] * w2[br * 49 + i * 7 + j];
            h2[j] = s;
        }
        for (int j = 0; j < 7; ++j) h[j] = h2[j];
        ln_relu7(h, g3 + br * 7, be3 + br * 7);
        for (int hd = 0; hd < 3; ++hd) {
            float s = b3[br * 3 + hd];
            for (int i = 0; i < 7; ++i) s += h[i] * w3[br * 21 + i * 3 + hd];
            mlp[br][off][hd] = s;
        }
    }
    __syncthreads();
    for (int idx = t; idx < 384; idx += 64) {
        int br = idx / 192, rem = idx % 192;
        int head = rem / 64, nm = rem % 64;
        int n = nm >> 3, m = nm & 7;
        int Hsp = br ? 4 : 2, Wsp = br ? 2 : 4;
        int nh = n / Wsp, nw = n % Wsp, mh = m / Wsp, mw = m % Wsp;
        int rel = (nh - mh + Hsp - 1) * (2 * Wsp - 1) + (nw - mw + Wsp - 1);
        btab[((br * 3 + head) * 8 + n) * 8 + m] = mlp[br][rel][head];
    }
}

// ---------------- 6. windowed attention (both branches), LDS-staged k/v ----------------

__global__ void attn_kernel(const float* __restrict__ qkv, const float* __restrict__ qp,
                            const float* __restrict__ btab, float* __restrict__ attb) {
    int branch = blockIdx.y;
    int flat = blockIdx.x;            // b*nW + wi, nW = 8192
    int b  = flat >> 13;
    int wi = flat & 8191;
    int Hsp = branch ? 4 : 2, Wsp = branch ? 2 : 4;
    int sh  = branch ? 2 : 1, sw  = branch ? 1 : 2;
    int WB = 256 / Wsp;
    int wy = wi / WB, wx = wi % WB;
    int qb = flat & 3;                // jnp.tile-vs-repeat quirk: qp batch = flat % B

    __shared__ float kv[2][8][60];    // [k|v][pos][ch]
    __shared__ int   regs[8];
    int t = threadIdx.x;

    for (int i = t; i < 960; i += 64) {
        int half = i / 480, rem = i % 480;
        int m = rem / 60, ch = rem % 60;
        int h = wy * Hsp + m / Wsp, w = wx * Wsp + m % Wsp;
        int oh = (h + sh) & 255, ow = (w + sw) & 255;
        kv[half][m][ch] = qkv[(((size_t)b << 16) + ((size_t)oh << 8) + ow) * 360 +
                              (half ? 240 : 120) + branch * 60 + ch];
    }
    if (t < 8) {
        int h = wy * Hsp + t / Wsp, w = wx * Wsp + t % Wsp;
        int hs = branch ? 4 : 2, hsh = branch ? 2 : 1;
        int ws = branch ? 2 : 4, wsh = branch ? 1 : 2;
        int rr = (h < 256 - hs) ? 0 : ((h < 256 - hsh) ? 1 : 2);
        int cr = (w < 256 - ws) ? 0 : ((w < 256 - wsh) ? 1 : 2);
        regs[t] = rr * 3 + cr;
    }
    __syncthreads();

    if (t < 24) {
        int head = t >> 3, n = t & 7;
        const float* qr = qp + ((((size_t)branch * 4 + qb) * 3 + head) * 8 + n) * 20;
        float q[20];
#pragma unroll
        for (int d = 0; d < 20; ++d) q[d] = qr[d] * 0.15811388300841897f; // (120/3)^-0.5
        float sc[8];
        float mx = -3.4e38f;
#pragma unroll
        for (int m = 0; m < 8; ++m) {
            float s = btab[((branch * 3 + head) * 8 + n) * 8 + m];
            if (regs[n] != regs[m]) s -= 100.f;
            const float* kr = &kv[0][m][head * 20];
#pragma unroll
            for (int d = 0; d < 20; ++d) s += q[d] * kr[d];
            sc[m] = s;
            mx = fmaxf(mx, s);
        }
        float den = 0.f;
#pragma unroll
        for (int m = 0; m < 8; ++m) { sc[m] = __expf(sc[m] - mx); den += sc[m]; }
        float inv = 1.f / den;
        int h = wy * Hsp + n / Wsp, w = wx * Wsp + n % Wsp;
        int oh = (h + sh) & 255, ow = (w + sw) & 255;   // roll back
        float* orow = attb + (((size_t)b << 16) + ((size_t)oh << 8) + ow) * 120 +
                      branch * 60 + head * 20;
#pragma unroll
        for (int d = 0; d < 20; ++d) {
            float o = 0.f;
#pragma unroll
            for (int m = 0; m < 8; ++m) o += sc[m] * kv[1][m][head * 20 + d];
            orow[d] = o * inv;
        }
    }
}

// ---------------- 7. depthwise 3x3 conv on V + residual add + bf16 pad ----------------

__global__ void conv_fuse(const float* __restrict__ qkv, const float* __restrict__ attb,
                          const float* __restrict__ cw, const float* __restrict__ cb,
                          unsigned short* __restrict__ Y) {
    size_t idx = (size_t)blockIdx.x * blockDim.x + threadIdx.x; // rows*128
    int    c   = (int)(idx & 127);
    size_t row = idx >> 7;
    if (c >= 120) { Y[idx] = 0; return; }
    int b = (int)(row >> 16);
    int pos = (int)(row & 65535);
    int h = pos >> 8, w = pos & 255;
    float acc = cb[c];
#pragma unroll
    for (int dy = -1; dy <= 1; ++dy) {
        int hh = h + dy;
        if ((unsigned)hh >= 256u) continue;
#pragma unroll
        for (int dx = -1; dx <= 1; ++dx) {
            int ww = w + dx;
            if ((unsigned)ww >= 256u) continue;
            acc += qkv[(((size_t)b << 16) + ((size_t)hh << 8) + ww) * 360 + 240 + c] *
                   cw[c * 9 + (dy + 1) * 3 + (dx + 1)];
        }
    }
    acc += attb[row * 120 + c];
    Y[idx] = f2bf(acc);
}

// ---------------- host launch ----------------

extern "C" void kernel_launch(void* const* d_in, const int* in_sizes, int n_in,
                              void* d_out, int out_size, void* d_ws, size_t ws_size,
                              hipStream_t stream) {
    (void)in_sizes; (void)n_in; (void)out_size; (void)ws_size;
    const float* x      = (const float*)d_in[0];
    // d_in[1], d_in[2] = H, W (ints; fixed at 256 here)
    const float* w_qkv  = (const float*)d_in[3];
    const float* b_qkv  = (const float*)d_in[4];
    const float* w_proj = (const float*)d_in[5];
    const float* b_proj = (const float*)d_in[6];
    const float* conv_w = (const float*)d_in[7];
    const float* conv_b = (const float*)d_in[8];
    const float* dp[14];
    for (int i = 0; i < 14; ++i) dp[i] = (const float*)d_in[9 + i];

    const size_t ROWS = 262144;               // B*L = 4*65536

    // workspace carve
    char* ws = (char*)d_ws;
    unsigned short* xb    = (unsigned short*)(ws);                     //  64 MB (reused for y)
    float*          qkv   = (float*)(ws + 67108864);                   // 360 MB
    float*          attb  = (float*)(ws + 67108864 + 377487360);       // 120 MB
    unsigned short* swzQ  = (unsigned short*)(ws + 570425344);         //  92 KB
    unsigned short* swzP  = (unsigned short*)(ws + 570519552);         //  32 KB
    float*          qp    = (float*)(ws + 570552320);                  //  15 KB
    float*          btab  = (float*)(ws + 570567680);                  // 1.5 KB

    // 1. x -> bf16 padded K=128
    cvt_pad128<<<(unsigned)((ROWS * 128 + 255) / 256), 256, 0, stream>>>(x, xb, ROWS * 128, 120);

    // 2. swizzle weights (w_qkv: 23 N-tiles; w_proj: 8 N-tiles; 4 K-tiles each)
    swizzle_w<<<(23 * 4 * 512 + 255) / 256, 256, 0, stream>>>(w_qkv, 120, 360, 23 * 4 * 512, swzQ);
    swizzle_w<<<(8  * 4 * 512 + 255) / 256, 256, 0, stream>>>(w_proj, 120, 120, 8 * 4 * 512, swzP);

    // 3. QKV GEMM: (262144 x 128) * (128 x 360) -> qkv f32  (A read once, B via LDS)
    wmma_gemm2<23><<<dim3(2048), 256, 0, stream>>>(xb, swzQ, b_qkv, qkv, 360, 360);

    // 4. pooled query
    pool_q<<<dim3(1920, 2), 256, 0, stream>>>(qkv, qp);

    // 5. dynamic position bias table
    dyn_bias<<<1, 64, 0, stream>>>(dp[0], dp[1], dp[2], dp[3], dp[4], dp[5], dp[6],
                                   dp[7], dp[8], dp[9], dp[10], dp[11], dp[12], dp[13], btab);

    // 6. windowed attention, both branches
    attn_kernel<<<dim3(32768, 2), 64, 0, stream>>>(qkv, qp, btab, attb);

    // 7. depthwise conv on V + residual + bf16 (reuse xb as y)
    conv_fuse<<<(unsigned)((ROWS * 128 + 255) / 256), 256, 0, stream>>>(qkv, attb, conv_w, conv_b, xb);

    // 8. output projection GEMM -> d_out (f32) with bias
    wmma_gemm2<8><<<dim3(2048), 256, 0, stream>>>(xb, swzP, b_proj, (float*)d_out, 120, 120);
}
